// Gemma4Attention_16724602650962
// MI455X (gfx1250) — compile-verified, matrix operand order checked by reading, not run on hardware
//
#include <hip/hip_runtime.h>

// ---------------------------------------------------------------------------
// Types for gfx1250 WMMA (wave32): v_wmma_f32_16x16x32_bf16
// ---------------------------------------------------------------------------
typedef __bf16 bf16;
typedef __attribute__((ext_vector_type(16))) __bf16 v16bf;
typedef __attribute__((ext_vector_type(8)))  __bf16 bf16x8;
typedef __attribute__((ext_vector_type(4)))  __bf16 bf16x4;
typedef __attribute__((ext_vector_type(8)))  float  v8f;

__device__ __forceinline__ v8f wmma_bf16(v16bf a, v16bf b, v8f c) {
  // (neg_a, A, neg_b, B, c_mod, C, reuse_a, reuse_b)
  return __builtin_amdgcn_wmma_f32_16x16x32_bf16(false, a, false, b,
                                                 (short)0, c, false, false);
}

// 16x32 bf16 fragment (A-layout; B 32x16 uses the same per-lane pattern with
// roles M<->N) from a row-major tile with 'stride' bf16 elements per row.
// lane<16 : row=lane,     K 0..7 then 16..23
// lane>=16: row=lane-16,  K 8..15 then 24..31
__device__ __forceinline__ v16bf load_frag(const bf16* __restrict__ base,
                                           int stride) {
  const int lane = threadIdx.x & 31;
  const int r    = lane & 15;
  const int sel  = lane >> 4;
  const bf16* p  = base + r * stride + sel * 8;
  bf16x8 lo = *(const bf16x8*)(p);
  bf16x8 hi = *(const bf16x8*)(p + 16);
  return __builtin_shufflevector(lo, hi, 0, 1, 2, 3, 4, 5, 6, 7,
                                         8, 9, 10, 11, 12, 13, 14, 15);
}

// ---------------------------------------------------------------------------
// Kernel 1/4: fp32 GEMM C[M,N] = A[M,K] * B[K,N] via bf16 WMMA.
// Block 256 threads (8 waves); tile 256x128, K-step 32. Each wave owns a
// 32x128 slab (2 A-frags x 8 B-frags = 16 WMMAs/k-step).
//  - global->register software pipelining (next K-tile loads overlap WMMAs)
//  - LDS double buffering: one barrier per k-step
//  - B-fragment register rotation: frag j+1 ds_loads issue before WMMAs of j
// B staged transposed so all fragment loads are contiguous b128 ds_loads.
// LDS row stride 40 (bank-conflict-free, 16B-aligned rows).
// ---------------------------------------------------------------------------
#define BM   256
#define BN   128
#define GLDT 40

__global__ __launch_bounds__(256)
void gemm_bf16_kernel(const float* __restrict__ A, const float* __restrict__ B,
                      float* __restrict__ C, int M, int N, int K) {
  __shared__ alignas(16) bf16 As[2][BM * GLDT];   // 2 x 20 KB
  __shared__ alignas(16) bf16 Bs[2][BN * GLDT];   // 2 x 10 KB

  const int tid  = threadIdx.x;
  const int wave = tid >> 5;
  const int lane = tid & 31;
  const int m0   = blockIdx.x * BM;
  const int n0   = blockIdx.y * BN;

  v8f acc[2][8];
#pragma unroll
  for (int h = 0; h < 2; ++h)
#pragma unroll
    for (int j = 0; j < 8; ++j) acc[h][j] = {};

  float4 pa[8], pb[4];
  auto load_stage = [&](int k0) {
#pragma unroll
    for (int i = 0; i < 8; ++i) {        // A tile: 256x32 = 2048 float4 chunks
      int c = tid + i * 256;
      int m = c >> 3, k = (c & 7) << 2;
      pa[i] = *(const float4*)(A + (size_t)(m0 + m) * K + k0 + k);
    }
#pragma unroll
    for (int i = 0; i < 4; ++i) {        // B tile: 32x128 = 1024 float4 chunks
      int c = tid + i * 256;
      int k = c >> 5, n = (c & 31) << 2;
      pb[i] = *(const float4*)(B + (size_t)(k0 + k) * N + n0 + n);
    }
  };
  auto store_stage = [&](int buf) {      // fp32 -> bf16 while storing to LDS
#pragma unroll
    for (int i = 0; i < 8; ++i) {
      int c = tid + i * 256;
      int m = c >> 3, k = (c & 7) << 2;
      bf16x4 h = {(bf16)pa[i].x, (bf16)pa[i].y, (bf16)pa[i].z, (bf16)pa[i].w};
      *(bf16x4*)(&As[buf][m * GLDT + k]) = h;
    }
#pragma unroll
    for (int i = 0; i < 4; ++i) {
      int c = tid + i * 256;
      int k = c >> 5, n = (c & 31) << 2;
      Bs[buf][(n + 0) * GLDT + k] = (bf16)pb[i].x;
      Bs[buf][(n + 1) * GLDT + k] = (bf16)pb[i].y;
      Bs[buf][(n + 2) * GLDT + k] = (bf16)pb[i].z;
      Bs[buf][(n + 3) * GLDT + k] = (bf16)pb[i].w;
    }
  };

  // prologue: fill buffer 0
  load_stage(0);
  store_stage(0);
  __syncthreads();

  const int iters = K >> 5;
  for (int it = 0; it < iters; ++it) {
    const int  buf  = it & 1;
    const bool more = (it + 1 < iters);
    if (more) load_stage((it + 1) << 5);           // overlap with compute
    if (it + 2 < iters) {                          // deepen the pipeline
      int k2 = (it + 2) << 5;
      __builtin_prefetch(A + (size_t)(m0 + (tid >> 1)) * K + k2, 0, 1);
      __builtin_prefetch(B + (size_t)(k2 + (tid >> 3)) * N + n0, 0, 1);
    }

    const bf16* as = &As[buf][0];
    const bf16* bs = &Bs[buf][0];
    v16bf a0 = load_frag(&as[(wave * 32 + 0)  * GLDT], GLDT);
    v16bf a1 = load_frag(&as[(wave * 32 + 16) * GLDT], GLDT);
    v16bf bnext = load_frag(&bs[0], GLDT);
#pragma unroll
    for (int j = 0; j < 8; ++j) {
      v16bf bcur = bnext;
      if (j < 7) bnext = load_frag(&bs[(j + 1) * 16 * GLDT], GLDT);
      acc[0][j] = wmma_bf16(a0, bcur, acc[0][j]);
      acc[1][j] = wmma_bf16(a1, bcur, acc[1][j]);
    }

    if (more) store_stage(buf ^ 1);                // fill the other buffer
    __syncthreads();                               // one barrier per k-step
  }

  // C layout: VGPR r -> rows r (lanes 0-15) and r+8 (lanes 16-31), col=lane&15
  const int col  = lane & 15;
  const int roff = (lane >> 4) << 3;
#pragma unroll
  for (int h = 0; h < 2; ++h)
#pragma unroll
    for (int j = 0; j < 8; ++j)
#pragma unroll
      for (int r = 0; r < 8; ++r) {
        int m = m0 + wave * 32 + h * 16 + r + roff;
        C[(size_t)m * N + n0 + j * 16 + col] = acc[h][j][r];
      }
}

// ---------------------------------------------------------------------------
// Kernel 2: per-(t,head) RMSNorm(q,k,v) + NeoX RoPE(q,k); emit bf16.
// ---------------------------------------------------------------------------
__global__ __launch_bounds__(256)
void normrope_kernel(const float* __restrict__ qkv,
                     const int* __restrict__ positions,
                     const float* __restrict__ qw, const float* __restrict__ kw,
                     bf16* __restrict__ qo, bf16* __restrict__ ko,
                     bf16* __restrict__ vo) {
  const int t = blockIdx.x, h = blockIdx.y, i = threadIdx.x;
  const int lane = i & 31, w = i >> 5;

  __shared__ float redq[8], redk[8], redv[8];
  __shared__ float qn_s[256], kn_s[256];

  const float* base = qkv + (size_t)t * 12288 + h * 256;
  float qv = base[i], kv = base[4096 + i], vv = base[8192 + i];

  float sq = qv * qv, sk = kv * kv, sv = vv * vv;
#pragma unroll
  for (int m = 16; m; m >>= 1) {
    sq += __shfl_xor(sq, m, 32);
    sk += __shfl_xor(sk, m, 32);
    sv += __shfl_xor(sv, m, 32);
  }
  if (lane == 0) { redq[w] = sq; redk[w] = sk; redv[w] = sv; }
  __syncthreads();
  float ssq = 0.f, ssk = 0.f, ssv = 0.f;
#pragma unroll
  for (int j = 0; j < 8; ++j) { ssq += redq[j]; ssk += redk[j]; ssv += redv[j]; }

  const float rq = rsqrtf(ssq * (1.f / 256.f) + 1e-6f);
  const float rk = rsqrtf(ssk * (1.f / 256.f) + 1e-6f);
  const float rv = rsqrtf(ssv * (1.f / 256.f) + 1e-6f);

  float qn = qv * rq * qw[i];
  float kn = kv * rk * kw[i];
  float vn = vv * rv;
  qn_s[i] = qn;
  kn_s[i] = kn;
  __syncthreads();

  const float pos = (float)positions[t];
  const int j = i & 127;
  const float freq = __powf(10000.f, -(2.f * (float)j) * (1.f / 256.f));
  const float ang = pos * freq;
  const float c = __cosf(ang), s = __sinf(ang);

  float q1 = qn_s[j], q2 = qn_s[j + 128];
  float k1 = kn_s[j], k2 = kn_s[j + 128];
  float outq = (i < 128) ? (q1 * c - q2 * s) : (q2 * c + q1 * s);
  float outk = (i < 128) ? (k1 * c - k2 * s) : (k2 * c + k1 * s);

  const size_t o = (size_t)t * 4096 + h * 256 + i;
  qo[o] = (bf16)outq;
  ko[o] = (bf16)outk;
  vo[o] = (bf16)vn;
}

// ---------------------------------------------------------------------------
// Kernel 3: causal flash attention, one head per blockIdx.y, 128 query rows
// per block (8 waves x 16 rows). All matmuls via v_wmma_f32_16x16x32_bf16.
// K/V tiles are register-prefetched one block ahead; B-fragments rotate
// through two register sets so ds_loads stay ahead of the WMMAs.
// ---------------------------------------------------------------------------
#define KSTR 264   // K-tile LDS row stride (bf16): 528B rows, conflict-free
#define VSTR 40    // transposed-V LDS row stride (80B rows, 16B aligned)
#define PSTR 40    // P staging row stride

__global__ __launch_bounds__(256)
void attn_kernel(const bf16* __restrict__ Q, const bf16* __restrict__ Kb,
                 const bf16* __restrict__ Vb, float* __restrict__ O) {
  __shared__ alignas(16) bf16 Ks[32 * KSTR];       // K tile, row-major [s][d]
  __shared__ alignas(16) bf16 Vt[256 * VSTR];      // V tile transposed [d][s]
  __shared__ alignas(16) bf16 Ps[8][16 * PSTR];    // per-wave P staging

  const int tid = threadIdx.x, lane = tid & 31, wave = tid >> 5;
  const int qb = blockIdx.x, h = blockIdx.y;
  const int qrow0 = qb * 128 + wave * 16;
  const size_t hoff = (size_t)h * 256;

  // ---- preload the wave's 16x256 Q tile as 8 A-fragments ----
  v16bf aQ[8];
  {
    const int r = lane & 15, sel = lane >> 4;
    const bf16* qp = Q + (size_t)(qrow0 + r) * 4096 + hoff + sel * 8;
#pragma unroll
    for (int dt = 0; dt < 8; ++dt) {
      bf16x8 lo = *(const bf16x8*)(qp + dt * 32);
      bf16x8 hi = *(const bf16x8*)(qp + dt * 32 + 16);
      aQ[dt] = __builtin_shufflevector(lo, hi, 0, 1, 2, 3, 4, 5, 6, 7,
                                               8, 9, 10, 11, 12, 13, 14, 15);
    }
  }

  float mi[8], li[8];
  v8f Oacc[16];
#pragma unroll
  for (int r = 0; r < 8; ++r) { mi[r] = -1e30f; li[r] = 0.f; }
#pragma unroll
  for (int t = 0; t < 16; ++t) Oacc[t] = {};

  // staging chunk geometry: c = tid + i*256 -> s = c>>5, d8 = c&31 (8 bf16)
  uint4 kx[4], vx[4];
  auto load_kv = [&](int s0) {
#pragma unroll
    for (int i = 0; i < 4; ++i) {
      int c = tid + i * 256;
      int s = c >> 5, d8 = c & 31;
      kx[i] = *(const uint4*)(Kb + (size_t)(s0 + s) * 4096 + hoff + d8 * 8);
      vx[i] = *(const uint4*)(Vb + (size_t)(s0 + s) * 4096 + hoff + d8 * 8);
    }
  };

  const int nblk = qb * 4 + 4;                 // 32-col K/V blocks, causal bound
  load_kv(0);
  for (int b = 0; b < nblk; ++b) {
    const int s0 = b * 32;

    // ---- store prefetched K tile (b128) and V tile (transposed scatter) ----
#pragma unroll
    for (int i = 0; i < 4; ++i) {
      int c = tid + i * 256;
      int s = c >> 5, d8 = c & 31;
      *(uint4*)(&Ks[s * KSTR + d8 * 8]) = kx[i];
      const bf16* e = (const bf16*)&vx[i];
#pragma unroll
      for (int j = 0; j < 8; ++j) Vt[(d8 * 8 + j) * VSTR + s] = e[j];
    }
    __syncthreads();

    if (b + 1 < nblk) load_kv(s0 + 32);        // overlap with compute

    // ---- S = Q * K^T  (two 16x16 score tiles, 8 WMMAs each) ----
    v8f S[2];
#pragma unroll
    for (int j = 0; j < 2; ++j) {
      v8f acc = {};
      v16bf bnext = load_frag(&Ks[j * 16 * KSTR], KSTR);
#pragma unroll
      for (int dt = 0; dt < 8; ++dt) {
        v16bf bcur = bnext;
        if (dt < 7) bnext = load_frag(&Ks[j * 16 * KSTR + (dt + 1) * 32], KSTR);
        acc = wmma_bf16(aQ[dt], bcur, acc);
      }
      const int col = s0 + j * 16 + (lane & 15);
      const int rb  = qrow0 + ((lane >> 4) << 3);
#pragma unroll
      for (int r = 0; r < 8; ++r)
        acc[r] = (col > rb + r) ? -3.0e38f : acc[r];
      S[j] = acc;
    }

    // ---- online softmax (rows live in 16-lane half-wave groups) ----
    float f[8];
#pragma unroll
    for (int r = 0; r < 8; ++r) {
      float mx = fmaxf(S[0][r], S[1][r]);
#pragma unroll
      for (int m = 1; m <= 8; m <<= 1) mx = fmaxf(mx, __shfl_xor(mx, m, 32));
      float mn = fmaxf(mi[r], mx);
      f[r] = __expf(mi[r] - mn);
      mi[r] = mn;
      float p0 = __expf(S[0][r] - mn);
      float p1 = __expf(S[1][r] - mn);
      S[0][r] = p0; S[1][r] = p1;
      float rs = p0 + p1;
#pragma unroll
      for (int m = 1; m <= 8; m <<= 1) rs += __shfl_xor(rs, m, 32);
      li[r] = li[r] * f[r] + rs;
    }
#pragma unroll
    for (int t = 0; t < 16; ++t)
#pragma unroll
      for (int r = 0; r < 8; ++r) Oacc[t][r] *= f[r];

    // ---- P: C-layout -> A-layout via wave-private LDS staging ----
    {
      const int n = lane & 15, half = lane >> 4;
      bf16* ps = &Ps[wave][0];
#pragma unroll
      for (int j = 0; j < 2; ++j)
#pragma unroll
        for (int r = 0; r < 8; ++r)
          ps[(r + half * 8) * PSTR + j * 16 + n] = (bf16)S[j][r];
    }
    v16bf aP = load_frag(&Ps[wave][0], PSTR);   // same-wave DS is in-order

    // ---- O += P * V  (16 d-tiles of 16x16, one WMMA each) ----
    {
      v16bf bnext = load_frag(&Vt[0], VSTR);
#pragma unroll
      for (int t = 0; t < 16; ++t) {
        v16bf bcur = bnext;
        if (t < 15) bnext = load_frag(&Vt[(t + 1) * 16 * VSTR], VSTR);
        Oacc[t] = wmma_bf16(aP, bcur, Oacc[t]);
      }
    }
    __syncthreads();
  }

  // ---- normalize and store fp32 attn output ----
  {
    const int n = lane & 15, half = lane >> 4;
#pragma unroll
    for (int r = 0; r < 8; ++r) {
      const float inv = 1.f / li[r];
      const int row = qrow0 + r + half * 8;
#pragma unroll
      for (int t = 0; t < 16; ++t)
        O[(size_t)row * 4096 + hoff + t * 16 + n] = Oacc[t][r] * inv;
    }
  }
}

// ---------------------------------------------------------------------------
// Host launcher
// ---------------------------------------------------------------------------
extern "C" void kernel_launch(void* const* d_in, const int* in_sizes, int n_in,
                              void* d_out, int out_size, void* d_ws,
                              size_t ws_size, hipStream_t stream) {
  const float* hidden    = (const float*)d_in[0];
  const int*   positions = (const int*)d_in[1];
  const float* w_qkv     = (const float*)d_in[2];
  const float* w_o       = (const float*)d_in[3];
  const float* q_norm_w  = (const float*)d_in[4];
  const float* k_norm_w  = (const float*)d_in[5];
  float*       out       = (float*)d_out;

  const size_t T = 2048;
  char* ws = (char*)d_ws;
  float* qkv = (float*)ws;                       // T x 12288 fp32 (96 MB)
  size_t off = T * 12288 * sizeof(float);
  bf16* qrot = (bf16*)(ws + off); off += T * 4096 * sizeof(bf16);
  bf16* krot = (bf16*)(ws + off); off += T * 4096 * sizeof(bf16);
  bf16* vnrm = (bf16*)(ws + off); off += T * 4096 * sizeof(bf16);
  float* attn = (float*)(ws + off);              // T x 4096 fp32 (32 MB)

  // 1) QKV projection: (2048x4096) @ (4096x12288)
  gemm_bf16_kernel<<<dim3(2048 / BM, 12288 / BN), 256, 0, stream>>>(
      hidden, w_qkv, qkv, 2048, 12288, 4096);

  // 2) RMSNorm + RoPE, emit bf16 q/k/v
  normrope_kernel<<<dim3(2048, 16), 256, 0, stream>>>(
      qkv, positions, q_norm_w, k_norm_w, qrot, krot, vnrm);

  // 3) Causal flash attention (16 q-row-blocks x 16 heads)
  attn_kernel<<<dim3(2048 / 128, 16), 256, 0, stream>>>(qrot, krot, vnrm, attn);

  // 4) Output projection: (2048x4096) @ (4096x4096)
  gemm_bf16_kernel<<<dim3(2048 / BM, 4096 / BN), 256, 0, stream>>>(
      attn, w_o, out, 2048, 4096, 4096);
}
